// LocalAttention_38783554683382
// MI455X (gfx1250) — matrix operand order repaired and checked by reading, not compiled, not run on hardware
//
#include <hip/hip_runtime.h>
#include <math.h>

typedef __attribute__((ext_vector_type(16))) _Float16 v16h;
typedef __attribute__((ext_vector_type(8)))  float    v8f;

#define WIN   128
#define DHEAD 64
#define NSEQ  4096
#define KSPAN 256                  // look_backward=1 -> 2*WIN keys
#define MASKVAL (-3.402823466e38f)
#define LOG2E   1.4426950408889634f
#define EXPBIAS 5.77f              // ~4/ln2: keeps exp(s) in f16 range, cancels in normalize

#define WMMA_F16(A, B, C) \
    __builtin_amdgcn_wmma_f32_16x16x32_f16(false, (A), false, (B), (short)0, (C), false, false)

// One 32-key chunk: sim tiles -> exp -> LDS relayout -> PV + ones-column row-sum.
template<bool MASKED>
__device__ __forceinline__ void attn_chunk(
    int jbase, int wave, int hs, int ln, bool t1_active,
    const _Float16 (*ks)[DHEAD + 4], const _Float16 (*vt)[KSPAN + 8],
    _Float16 (*pst)[34],
    const v16h& a0, const v16h& a1, const v16h& ones,
    v8f& o0, v8f& o1, v8f& o2, v8f& o3, v8f& o4)
{
    v8f c0 = {}, c1 = {};
    // ---- QK^T tile 0 ----
    {
        v16h b0, b1;
        #pragma unroll
        for (int e = 0; e < 16; ++e) {
            int kk2 = hs * 16 + e;                 // B layout: K split by lane half
            b0[e] = ks[jbase + ln][kk2];
            b1[e] = ks[jbase + ln][32 + kk2];
        }
        c0 = WMMA_F16(a0, b0, c0);
        c0 = WMMA_F16(a1, b1, c0);
    }
    // ---- QK^T tile 1 (skipped when fully above the diagonal) ----
    if (t1_active) {
        v16h b0, b1;
        #pragma unroll
        for (int e = 0; e < 16; ++e) {
            int kk2 = hs * 16 + e;
            b0[e] = ks[jbase + 16 + ln][kk2];
            b1[e] = ks[jbase + 16 + ln][32 + kk2];
        }
        c1 = WMMA_F16(a0, b0, c1);
        c1 = WMMA_F16(a1, b1, c1);
    }

    // ---- exp (no max subtraction: |sim| ~ O(6); constant bias folded into fma) ----
    #pragma unroll
    for (int vv = 0; vv < 8; ++vv) {
        float s0, s1;
        if (MASKED) {
            int allow = wave * 16 + vv + 8 * hs + 128;   // max attendable key col
            s0 = (jbase + ln      > allow) ? MASKVAL : c0[vv];
            s1 = (!t1_active || (jbase + 16 + ln > allow)) ? MASKVAL : c1[vv];
        } else {
            s0 = c0[vv];
            s1 = c1[vv];
        }
        // exp(s)*2^-EXPBIAS; masked: fma overflows to -inf -> exp2 -> exactly 0
        float p0 = __builtin_amdgcn_exp2f(__builtin_fmaf(s0, LOG2E, -EXPBIAS));
        float p1 = __builtin_amdgcn_exp2f(__builtin_fmaf(s1, LOG2E, -EXPBIAS));
        pst[vv + 8 * hs][ln]      = (_Float16)p0;
        pst[vv + 8 * hs][16 + ln] = (_Float16)p1;
    }

    // ---- re-layout probs (C-layout -> A-fragment) via per-wave LDS tile ----
    v16h pf;
    #pragma unroll
    for (int e = 0; e < 16; ++e) {
        int kk = hs * 16 + e;
        pf[e] = pst[ln][kk];
    }

    // ---- P x V: 4 output tiles + ones-column tile (row sums via WMMA) ----
    #pragma unroll
    for (int t = 0; t < 4; ++t) {
        v16h b;
        #pragma unroll
        for (int e = 0; e < 16; ++e) {
            int kk2 = hs * 16 + e;
            b[e] = vt[t * 16 + ln][jbase + kk2];
        }
        if      (t == 0) o0 = WMMA_F16(pf, b, o0);
        else if (t == 1) o1 = WMMA_F16(pf, b, o1);
        else if (t == 2) o2 = WMMA_F16(pf, b, o2);
        else             o3 = WMMA_F16(pf, b, o3);
    }
    o4 = WMMA_F16(pf, ones, o4);   // every column of o4 = sum_k p[row][k]
}

__global__ __launch_bounds__(256)
void local_attn_rope_kernel(const float* __restrict__ q,
                            const float* __restrict__ k,
                            const float* __restrict__ v,
                            float* __restrict__ out)
{
    __shared__ float    invf[32];                  // RoPE inverse frequencies
    __shared__ _Float16 qs[WIN][DHEAD + 4];        // rotated queries * D^-0.5
    __shared__ _Float16 ks[KSPAN][DHEAD + 4];      // rotated keys (+look-back, pad=-1)
    __shared__ _Float16 vt[DHEAD][KSPAN + 8];      // values, transposed
    __shared__ _Float16 pstage[8][16][34];         // per-wave attn-prob relayout tile

    const int win = blockIdx.x;
    const int bh  = blockIdx.y;
    const int tid = threadIdx.x;

    const float* qg = q   + ((size_t)bh * NSEQ + (size_t)win * WIN) * DHEAD;
    const float* kg = k   + (size_t)bh * NSEQ * DHEAD;
    const float* vg = v   + (size_t)bh * NSEQ * DHEAD;
    float*       og = out + ((size_t)bh * NSEQ + (size_t)win * WIN) * DHEAD;

    const float LN1E4_D32 = 0.28782313662425572f;   // ln(10000)/32
    const float SCALE     = 0.125f;                 // D^-0.5, folded into q

    if (tid < 32) invf[tid] = __expf(-(float)tid * LN1E4_D32);
    __syncthreads();

    // ---- load + RoPE queries (float4 per half), pre-scaled ----
    for (int idx = tid; idx < WIN * 8; idx += 256) {
        int row = idx >> 3, g = (idx & 7) << 2;     // 4 consecutive dims of each half
        float4 x1 = *(const float4*)(qg + row * DHEAD + g);
        float4 x2 = *(const float4*)(qg + row * DHEAD + g + 32);
        float t = (float)(win * WIN + row);
        float a1c[4] = {x1.x, x1.y, x1.z, x1.w};
        float a2c[4] = {x2.x, x2.y, x2.z, x2.w};
        #pragma unroll
        for (int c = 0; c < 4; ++c) {
            float f = t * invf[g + c];
            float sf, cf; __sincosf(f, &sf, &cf);
            qs[row][g + c]      = (_Float16)((a1c[c] * cf - a2c[c] * sf) * SCALE);
            qs[row][g + c + 32] = (_Float16)((a2c[c] * cf + a1c[c] * sf) * SCALE);
        }
    }
    // ---- load + RoPE keys with look-back; out-of-range -> -1 (reference pad) ----
    for (int idx = tid; idx < KSPAN * 8; idx += 256) {
        int j = idx >> 3, g = (idx & 7) << 2;
        int gk = win * WIN - WIN + j;
        if (gk < 0) {
            #pragma unroll
            for (int c = 0; c < 4; ++c) {
                ks[j][g + c]      = (_Float16)(-1.0f);
                ks[j][g + c + 32] = (_Float16)(-1.0f);
            }
        } else {
            float4 x1 = *(const float4*)(kg + (size_t)gk * DHEAD + g);
            float4 x2 = *(const float4*)(kg + (size_t)gk * DHEAD + g + 32);
            float t = (float)gk;
            float a1c[4] = {x1.x, x1.y, x1.z, x1.w};
            float a2c[4] = {x2.x, x2.y, x2.z, x2.w};
            #pragma unroll
            for (int c = 0; c < 4; ++c) {
                float f = t * invf[g + c];
                float sf, cf; __sincosf(f, &sf, &cf);
                ks[j][g + c]      = (_Float16)(a1c[c] * cf - a2c[c] * sf);
                ks[j][g + c + 32] = (_Float16)(a2c[c] * cf + a1c[c] * sf);
            }
        }
    }
    // ---- load values transposed (float4); pad=-1 ----
    for (int idx = tid; idx < KSPAN * 16; idx += 256) {
        int j = idx >> 4, d = (idx & 15) << 2;
        int gk = win * WIN - WIN + j;
        float4 val = make_float4(-1.0f, -1.0f, -1.0f, -1.0f);
        if (gk >= 0) val = *(const float4*)(vg + (size_t)gk * DHEAD + d);
        vt[d + 0][j] = (_Float16)val.x;
        vt[d + 1][j] = (_Float16)val.y;
        vt[d + 2][j] = (_Float16)val.z;
        vt[d + 3][j] = (_Float16)val.w;
    }
    __syncthreads();

    const int wave = tid >> 5;
    const int lane = tid & 31;
    const int hs   = lane >> 4;   // lane-half select
    const int ln   = lane & 15;

    // A fragments for this wave's 16 query rows (contiguous K split by lane half, matches B)
    v16h a0, a1;
    #pragma unroll
    for (int e = 0; e < 16; ++e) {
        int kk = hs * 16 + e;   // 0..31
        a0[e] = qs[wave * 16 + ln][kk];
        a1[e] = qs[wave * 16 + ln][32 + kk];
    }

    // Constant all-ones B fragment (register-resident) for row-sum WMMA
    v16h ones;
    #pragma unroll
    for (int e = 0; e < 16; ++e) ones[e] = (_Float16)1.0f;

    v8f o0 = {}, o1 = {}, o2 = {}, o3 = {}, o4 = {};

    // Chunks fully below the causal diagonal: jbase <= 16*wave + 97
    const int jc_clear = ((wave * 16 + 97) >> 5) + 1;
    // Last chunk containing any attendable key: jbase <= 16*wave + 143
    const int jc_max   = ((wave * 16 + 143) >> 5) + 1;

    for (int jc = 0; jc < jc_clear; ++jc)
        attn_chunk<false>(jc * 32, wave, hs, ln, true,
                          ks, vt, pstage[wave], a0, a1, ones, o0, o1, o2, o3, o4);

    for (int jc = jc_clear; jc < jc_max; ++jc) {
        bool t1_active = (jc * 32 + 16) <= (wave * 16 + 143);  // wave-uniform
        attn_chunk<true>(jc * 32, wave, hs, ln, t1_active,
                         ks, vt, pstage[wave], a0, a1, ones, o0, o1, o2, o3, o4);
    }

    // ---- normalize + store (C-layout: row = vv + 8*hs, col = ln + 16*t) ----
    #pragma unroll
    for (int vv = 0; vv < 8; ++vv) {
        int   row = wave * 16 + vv + 8 * hs;
        float ivl = 1.0f / o4[vv];     // row sum of probs (bias cancels)
        og[row * DHEAD +  0 + ln] = o0[vv] * ivl;
        og[row * DHEAD + 16 + ln] = o1[vv] * ivl;
        og[row * DHEAD + 32 + ln] = o2[vv] * ivl;
        og[row * DHEAD + 48 + ln] = o3[vv] * ivl;
    }
}

extern "C" void kernel_launch(void* const* d_in, const int* in_sizes, int n_in,
                              void* d_out, int out_size, void* d_ws, size_t ws_size,
                              hipStream_t stream) {
    const float* q = (const float*)d_in[0];
    const float* k = (const float*)d_in[1];
    const float* v = (const float*)d_in[2];
    float* out = (float*)d_out;
    const int bh = in_sizes[0] / (NSEQ * DHEAD);    // B*H = 64
    dim3 grid(NSEQ / WIN, bh);                      // 32 windows x 64 heads
    local_attn_rope_kernel<<<grid, 256, 0, stream>>>(q, k, v, out);
}
